// TransformerBlock_41343355191929
// MI455X (gfx1250) — matrix operand order repaired and checked by reading, not compiled, not run on hardware
//
#include <hip/hip_runtime.h>
#include <math.h>

typedef _Float16 f16;
typedef _Float16 v8h  __attribute__((ext_vector_type(8)));
typedef _Float16 v16h __attribute__((ext_vector_type(16)));
typedef float    v8f  __attribute__((ext_vector_type(8)));

#define H_DIM   1024
#define SEQ     2048
#define BATCH   2
#define NHEADS  16
#define HEADDIM 64
#define ROWS    (BATCH*SEQ)   // 4096 token rows

__device__ __forceinline__ v8f wmma_f16(v16h a, v16h b, v8f c) {
    return __builtin_amdgcn_wmma_f32_16x16x32_f16(
        false, a, false, b, (short)0, c, false, false);
}

__device__ __forceinline__ v16h cat8(v8h a, v8h b) {
    return __builtin_shufflevector(a, b, 0,1,2,3,4,5,6,7,8,9,10,11,12,13,14,15);
}

__device__ __forceinline__ v8f v8f_zero() {
    v8f z;
    #pragma unroll
    for (int i = 0; i < 8; ++i) z[i] = 0.0f;
    return z;
}

// CDNA5 async global->LDS copy (ASYNCcnt path, no VGPR round trip).
// LDS flat addresses: low 32 bits are the LDS byte offset (ISA 10.2 aperture).
__device__ __forceinline__ void async_copy_b128(f16* lds_dst, const f16* gsrc) {
    unsigned laddr = (unsigned)(unsigned long long)lds_dst;
    asm volatile("global_load_async_to_lds_b128 %0, %1, off"
                 :: "v"(laddr), "v"(gsrc) : "memory");
}
#define WAIT_ASYNC(n) asm volatile("s_wait_asynccnt %0" :: "n"(n) : "memory")

// ---------------------------------------------------------------------------
// Weight prep: f32 [K][N] -> f16 [N][K] (transpose so GEMM B stages along K)
// 32x32 tiles via LDS; 256 threads.
// ---------------------------------------------------------------------------
__global__ __launch_bounds__(256) void k_convert_transpose(
    const float* __restrict__ src, f16* __restrict__ dst, int K, int N)
{
    __shared__ float t[32][33];
    const int kb = blockIdx.x * 32, nb = blockIdx.y * 32;
    const int tx = threadIdx.x & 31, ty = threadIdx.x >> 5;   // ty: 0..7
    #pragma unroll
    for (int r = ty; r < 32; r += 8)
        t[r][tx] = src[(size_t)(kb + r) * N + nb + tx];
    __syncthreads();
    #pragma unroll
    for (int r = ty; r < 32; r += 8)
        dst[(size_t)(nb + r) * K + kb + tx] = (f16)t[tx][r];
}

// ---------------------------------------------------------------------------
// LayerNorm over last dim, fp32 in -> fp16 out; one block (256 thr) per row
// ---------------------------------------------------------------------------
__global__ __launch_bounds__(256) void k_layernorm_f16(
    const float* __restrict__ x, const float* __restrict__ g,
    const float* __restrict__ b, f16* __restrict__ out, int ncols)
{
    __shared__ float red[256];
    const int row = blockIdx.x;
    const int tid = threadIdx.x;
    const float* xr = x + (size_t)row * ncols;

    float s = 0.f;
    for (int c = tid; c < ncols; c += 256) s += xr[c];
    red[tid] = s; __syncthreads();
    for (int o = 128; o > 0; o >>= 1) {
        if (tid < o) red[tid] += red[tid + o];
        __syncthreads();
    }
    const float mu = red[0] / (float)ncols;
    __syncthreads();

    float v = 0.f;
    for (int c = tid; c < ncols; c += 256) { float d = xr[c] - mu; v += d * d; }
    red[tid] = v; __syncthreads();
    for (int o = 128; o > 0; o >>= 1) {
        if (tid < o) red[tid] += red[tid + o];
        __syncthreads();
    }
    const float rstd = rsqrtf(red[0] / (float)ncols + 1e-5f);

    f16* outr = out + (size_t)row * ncols;
    for (int c = tid; c < ncols; c += 256)
        outr[c] = (f16)((xr[c] - mu) * rstd * g[c] + b[c]);
}

// ---------------------------------------------------------------------------
// WMMA GEMM: C[M,N] = A[M,K](f16,rm) @ Bt[N,K](f16, pre-transposed) + bias
// flags: 1 = +bias, 2 = exact GELU, 4 = +resid (f32 [M,N])
// Block 128 threads = 4 waves; tile 128x64, BK=32.
// Both tiles staged with async b128 copies into double-buffered LDS;
// every fragment is two ds_load_b128. 8 WMMA per wave per K-step.
// ---------------------------------------------------------------------------
#define BM 128
#define BN 64
#define BK 32
#define SPAD 40   // 32+8 halves row stride = 80 B (multiple of 16 B)

__global__ __launch_bounds__(128) void k_gemm_f16(
    const f16* __restrict__ A, const f16* __restrict__ Bt,
    const float* __restrict__ bias, const float* __restrict__ resid,
    float* __restrict__ Cf32, f16* __restrict__ Cf16,
    int M, int N, int K, int flags)
{
    __shared__ __align__(16) f16 sA[2][BM][SPAD];    // (m,k) double-buffered
    __shared__ __align__(16) f16 sBT[2][BN][SPAD];   // (n,k) double-buffered

    const int tid  = threadIdx.x;
    const int wave = tid >> 5;
    const int lane = tid & 31;
    const int lo   = (lane < 16) ? 1 : 0;
    const int ln   = lane & 15;
    const int tileM = blockIdx.y * BM;
    const int tileN = blockIdx.x * BN;

    // per-thread staging coordinates (chunk = 8 halves)
    const int srow = tid >> 2;            // 0..31
    const int scol = (tid & 3) << 3;      // 0,8,16,24

    // issue 6 async b128 copies: 4 for A rows, 2 for Bt rows
    auto stage = [&](int kt, int buf) {
        const int k0 = kt * BK;
        const f16* ga = A + (size_t)(tileM + srow) * K + k0 + scol;
        f16* la = &sA[buf][srow][scol];
        #pragma unroll
        for (int u = 0; u < 4; ++u) {
            async_copy_b128(la, ga);
            ga += (size_t)32 * K;
            la += 32 * SPAD;
        }
        const f16* gb = Bt + (size_t)(tileN + srow) * K + k0 + scol;
        f16* lb = &sBT[buf][srow][scol];
        #pragma unroll
        for (int u = 0; u < 2; ++u) {
            async_copy_b128(lb, gb);
            gb += (size_t)32 * K;
            lb += 32 * SPAD;
        }
    };

    v8f acc0[4], acc1[4];
    #pragma unroll
    for (int j = 0; j < 4; ++j) { acc0[j] = v8f_zero(); acc1[j] = v8f_zero(); }

    const int nK = K / BK;
    stage(0, 0);
    for (int kt = 0; kt < nK; ++kt) {
        const int buf = kt & 1;
        if (kt + 1 < nK) { stage(kt + 1, buf ^ 1); WAIT_ASYNC(6); }
        else             { WAIT_ASYNC(0); }
        __syncthreads();   // current tile visible to all waves

        const f16* arow0 = &sA[buf][32 * wave + ln][lo ? 0 : 8];
        const f16* arow1 = arow0 + 16 * SPAD;
        const v16h a0 = cat8(*(const v8h*)arow0, *(const v8h*)(arow0 + 16));
        const v16h a1 = cat8(*(const v8h*)arow1, *(const v8h*)(arow1 + 16));
        #pragma unroll
        for (int j = 0; j < 4; ++j) {
            const f16* brow = &sBT[buf][j * 16 + ln][lo ? 0 : 8];
            const v16h bf = cat8(*(const v8h*)brow, *(const v8h*)(brow + 16));
            acc0[j] = wmma_f16(a0, bf, acc0[j]);
            acc1[j] = wmma_f16(a1, bf, acc1[j]);
        }
        __syncthreads();   // reads done before this buffer is restaged
    }

    // ---- epilogue: C layout row = i + (lo?0:8), col = ln ----
    #pragma unroll
    for (int t = 0; t < 2; ++t) {
        #pragma unroll
        for (int j = 0; j < 4; ++j) {
            const int col = tileN + j * 16 + ln;
            const float bv = (flags & 1) ? bias[col] : 0.0f;
            const v8f* accp = t ? &acc1[j] : &acc0[j];
            #pragma unroll
            for (int i = 0; i < 8; ++i) {
                const int row = tileM + 32 * wave + 16 * t + i + (lo ? 0 : 8);
                float v = (*accp)[i] + bv;
                if (flags & 2) v = 0.5f * v * (1.0f + erff(v * 0.70710678118654752f));
                if (flags & 4) v += resid[(size_t)row * N + col];
                if (Cf32) Cf32[(size_t)row * N + col] = v;
                if (Cf16) Cf16[(size_t)row * N + col] = (f16)v;
            }
        }
    }
}

// ---------------------------------------------------------------------------
// Fused flash-style attention. qkv: f16 [ROWS, 3*H] (q|k|v concatenated).
// Block 128 threads = 4 waves on one (b,head); each wave owns 16 query rows.
// K fragments: direct b128 global loads (hd contiguous).
// V tile: cooperatively staged transposed in LDS (pointer-increment loop).
// ---------------------------------------------------------------------------
__global__ __launch_bounds__(128) void k_attention(
    const f16* __restrict__ qkv, f16* __restrict__ out, float scale)
{
    __shared__ __align__(16) f16 sVT[HEADDIM][SPAD];   // (hd, key)
    __shared__ __align__(16) f16 sP[4][16][32];        // per-wave P tile

    const int tid  = threadIdx.x;
    const int wave = tid >> 5;
    const int lane = tid & 31;
    const int lo   = (lane < 16) ? 1 : 0;
    const int ln   = lane & 15;

    const int bh   = blockIdx.y;
    const int b    = bh / NHEADS;
    const int head = bh % NHEADS;
    const int q0   = blockIdx.x * 64 + wave * 16;
    const int LD   = 3 * H_DIM;

    const size_t base = (size_t)b * SEQ * LD;
    const f16* qb = qkv + base + head * HEADDIM;
    const f16* kb = qkv + base + H_DIM + head * HEADDIM;
    const f16* vb = qkv + base + 2 * H_DIM + head * HEADDIM;

    // Q fragments: two b128 global loads each
    v16h aq[2];
    #pragma unroll
    for (int st = 0; st < 2; ++st) {
        const f16* qr = qb + (size_t)(q0 + ln) * LD + 32 * st + (lo ? 0 : 8);
        aq[st] = cat8(*(const v8h*)qr, *(const v8h*)(qr + 16));
    }

    v8f o[4];
    #pragma unroll
    for (int j = 0; j < 4; ++j) o[j] = v8f_zero();
    float m[8], l[8];
    #pragma unroll
    for (int i = 0; i < 8; ++i) { m[i] = -1e30f; l[i] = 0.0f; }

    // per-thread V-staging coordinates: key r0 = tid>>5 (+4/iter), hd pair c
    const int vr0 = tid >> 5;
    const int vc  = (tid & 31) << 1;

    for (int kt = 0; kt < SEQ / 32; ++kt) {
        const int kk0 = kt * 32;

        // ---- cooperative stage of V tile, transposed; exact 8 iterations ----
        {
            const f16* vsrc = vb + (size_t)(kk0 + vr0) * LD + vc;
            f16* vdst = &sVT[vc][vr0];
            #pragma unroll
            for (int u = 0; u < 8; ++u) {
                vdst[0]    = vsrc[0];
                vdst[SPAD] = vsrc[1];
                vsrc += (size_t)4 * LD;
                vdst += 4;
            }
        }

        // ---- scores: S[16q x 32k] = Q @ K^T; K frags via b128 global loads ----
        v8f s0 = v8f_zero(), s1 = v8f_zero();
        #pragma unroll
        for (int st = 0; st < 2; ++st) {
            const f16* kr0 = kb + (size_t)(kk0 + ln) * LD      + 32 * st + (lo ? 0 : 8);
            const f16* kr1 = kb + (size_t)(kk0 + 16 + ln) * LD + 32 * st + (lo ? 0 : 8);
            const v16h bk0 = cat8(*(const v8h*)kr0, *(const v8h*)(kr0 + 16));
            const v16h bk1 = cat8(*(const v8h*)kr1, *(const v8h*)(kr1 + 16));
            s0 = wmma_f16(aq[st], bk0, s0);
            s1 = wmma_f16(aq[st], bk1, s1);
        }

        // ---- online softmax (rows spread across 16-lane halves) ----
        #pragma unroll
        for (int i = 0; i < 8; ++i) {
            float a0 = s0[i] * scale, a1 = s1[i] * scale;
            float rm = fmaxf(a0, a1);
            #pragma unroll
            for (int off = 1; off < 16; off <<= 1)
                rm = fmaxf(rm, __shfl_xor(rm, off, 32));
            float mn    = fmaxf(m[i], rm);
            float alpha = __expf(m[i] - mn);
            float p0 = __expf(a0 - mn), p1 = __expf(a1 - mn);
            float rs = p0 + p1;
            #pragma unroll
            for (int off = 1; off < 16; off <<= 1)
                rs += __shfl_xor(rs, off, 32);
            l[i] = l[i] * alpha + rs;
            m[i] = mn;
            #pragma unroll
            for (int j = 0; j < 4; ++j) o[j][i] *= alpha;
            const int r = i + (lo ? 0 : 8);
            sP[wave][r][ln]      = (f16)p0;
            sP[wave][r][16 + ln] = (f16)p1;
        }
        __syncthreads();   // V-tile + P stores visible

        // ---- P as A fragment: two b128 LDS loads ----
        const f16* prow = &sP[wave][ln][lo ? 0 : 8];
        const v16h ap = cat8(*(const v8h*)prow, *(const v8h*)(prow + 16));
        // ---- O += P @ V: V frags from transposed LDS, two b128 each ----
        #pragma unroll
        for (int j = 0; j < 4; ++j) {
            const f16* vrow = &sVT[j * 16 + ln][lo ? 0 : 8];
            const v16h bv = cat8(*(const v8h*)vrow, *(const v8h*)(vrow + 16));
            o[j] = wmma_f16(ap, bv, o[j]);
        }
        __syncthreads();   // reads done before next tile overwrites LDS
    }

    // ---- finalize: O / l, write f16 [ROWS, H] (heads concatenated) ----
    #pragma unroll
    for (int j = 0; j < 4; ++j) {
        #pragma unroll
        for (int i = 0; i < 8; ++i) {
            const int row = q0 + i + (lo ? 0 : 8);
            const int col = head * HEADDIM + j * 16 + ln;
            out[(size_t)(b * SEQ + row) * H_DIM + col] = (f16)(o[j][i] / l[i]);
        }
    }
}

// ---------------------------------------------------------------------------
// Launch sequence
// ---------------------------------------------------------------------------
extern "C" void kernel_launch(void* const* d_in, const int* in_sizes, int n_in,
                              void* d_out, int out_size, void* d_ws, size_t ws_size,
                              hipStream_t stream) {
    (void)in_sizes; (void)n_in; (void)out_size; (void)ws_size;
    const float* x    = (const float*)d_in[0];
    const float* ln1g = (const float*)d_in[1];
    const float* ln1b = (const float*)d_in[2];
    const float* Wqkv = (const float*)d_in[3];
    const float* bqkv = (const float*)d_in[4];
    const float* Wout = (const float*)d_in[5];
    const float* bout = (const float*)d_in[6];
    const float* ln2g = (const float*)d_in[7];
    const float* ln2b = (const float*)d_in[8];
    const float* W1   = (const float*)d_in[9];
    const float* b1   = (const float*)d_in[10];
    const float* W2   = (const float*)d_in[11];
    const float* b2   = (const float*)d_in[12];
    float* out = (float*)d_out;

    char* ws = (char*)d_ws;
    size_t off = 0;
    auto alloc = [&](size_t bytes) -> void* {
        void* p = ws + off;
        off += (bytes + 255) & ~(size_t)255;
        return p;
    };
    f16*   h1     = (f16*)  alloc((size_t)ROWS * H_DIM * 2);
    f16*   qkv16  = (f16*)  alloc((size_t)ROWS * 3 * H_DIM * 2);
    f16*   attn   = (f16*)  alloc((size_t)ROWS * H_DIM * 2);
    float* x2     = (float*)alloc((size_t)ROWS * H_DIM * 4);
    f16*   h2     = (f16*)  alloc((size_t)ROWS * H_DIM * 2);
    f16*   ffn1   = (f16*)  alloc((size_t)ROWS * 4 * H_DIM * 2);
    f16*   WqkvT  = (f16*)  alloc((size_t)H_DIM * 3 * H_DIM * 2);   // [3H][H]
    f16*   WoutT  = (f16*)  alloc((size_t)H_DIM * H_DIM * 2);       // [H][H]
    f16*   W1T    = (f16*)  alloc((size_t)H_DIM * 4 * H_DIM * 2);   // [4H][H]
    f16*   W2T    = (f16*)  alloc((size_t)4 * H_DIM * H_DIM * 2);   // [H][4H]

    // weights: f32 [K][N] -> f16 [N][K]
    k_convert_transpose<<<dim3(H_DIM / 32, 3 * H_DIM / 32), 256, 0, stream>>>(
        Wqkv, WqkvT, H_DIM, 3 * H_DIM);
    k_convert_transpose<<<dim3(H_DIM / 32, H_DIM / 32), 256, 0, stream>>>(
        Wout, WoutT, H_DIM, H_DIM);
    k_convert_transpose<<<dim3(H_DIM / 32, 4 * H_DIM / 32), 256, 0, stream>>>(
        W1, W1T, H_DIM, 4 * H_DIM);
    k_convert_transpose<<<dim3(4 * H_DIM / 32, H_DIM / 32), 256, 0, stream>>>(
        W2, W2T, 4 * H_DIM, H_DIM);

    k_layernorm_f16<<<ROWS, 256, 0, stream>>>(x, ln1g, ln1b, h1, H_DIM);

    k_gemm_f16<<<dim3(3 * H_DIM / BN, ROWS / BM), 128, 0, stream>>>(
        h1, WqkvT, bqkv, nullptr, nullptr, qkv16, ROWS, 3 * H_DIM, H_DIM, 1);

    k_attention<<<dim3(SEQ / 64, BATCH * NHEADS), 128, 0, stream>>>(
        qkv16, attn, 0.125f);

    k_gemm_f16<<<dim3(H_DIM / BN, ROWS / BM), 128, 0, stream>>>(
        attn, WoutT, bout, x, x2, nullptr, ROWS, H_DIM, H_DIM, 1 | 4);

    k_layernorm_f16<<<ROWS, 256, 0, stream>>>(x2, ln2g, ln2b, h2, H_DIM);

    k_gemm_f16<<<dim3(4 * H_DIM / BN, ROWS / BM), 128, 0, stream>>>(
        h2, W1T, b1, nullptr, nullptr, ffn1, ROWS, 4 * H_DIM, H_DIM, 1 | 2);

    k_gemm_f16<<<dim3(H_DIM / BN, ROWS / BM), 128, 0, stream>>>(
        ffn1, W2T, b2, x2, out, nullptr, ROWS, H_DIM, 4 * H_DIM, 1 | 4);
}
